// NeuralBP_27590869909731
// MI455X (gfx1250) — 16-bit and 8-bit matrix operand order repaired and checked by reading, compiled, not run on hardware
//
#include <hip/hip_runtime.h>
#include <hip/hip_bf16.h>


#define NMAT 4096

typedef __attribute__((ext_vector_type(16))) _Float16 v16h;
typedef __attribute__((ext_vector_type(8)))  _Float16 v8h;
typedef __attribute__((ext_vector_type(8)))  float    v8f;
typedef __attribute__((ext_vector_type(16))) int      v16i;

#if defined(__has_builtin)
#if __has_builtin(__builtin_amdgcn_global_load_async_to_lds_b128) && \
    __has_builtin(__builtin_amdgcn_s_wait_asynccnt)
#define USE_ASYNC_LDS 1
#endif
#endif
#ifndef USE_ASYNC_LDS
#define USE_ASYNC_LDS 0
#endif

typedef int v4i __attribute__((vector_size(16)));
typedef __attribute__((address_space(1))) v4i GA_v4i;
typedef __attribute__((address_space(3))) v4i LA_v4i;

__device__ __forceinline__ void copy16(const void* g, void* l) {
#if USE_ASYNC_LDS
    __builtin_amdgcn_global_load_async_to_lds_b128((GA_v4i*)g, (LA_v4i*)l, 0, 0);
#else
    *(int4*)l = *(const int4*)g;
#endif
}
__device__ __forceinline__ void waitCopies() {
#if USE_ASYNC_LDS
    __builtin_amdgcn_s_wait_asynccnt(0);
#endif
}

__device__ __forceinline__ float signf(float x) {
    return (x > 0.f) ? 1.f : ((x < 0.f) ? -1.f : 0.f);
}
__device__ __forceinline__ v16h mk16(v8h lo, v8h hi) {
    union { v16h v; v8h h[2]; } u;
    u.h[0] = lo; u.h[1] = hi;
    return u.v;
}

__global__ __launch_bounds__(256) void prepH(const float* __restrict__ H,
                                             unsigned char* __restrict__ Hf8,
                                             _Float16* __restrict__ HfT) {
    __shared__ _Float16 t[32][33];
    const int bx = blockIdx.x * 32, by = blockIdx.y * 32;
    const int tx = threadIdx.x & 31, ty = threadIdx.x >> 5;
    #pragma unroll
    for (int r = 0; r < 4; ++r) {
        const int row = by + ty + 8 * r, col = bx + tx;
        const float h = H[(size_t)row * NMAT + col];
        Hf8[(size_t)row * NMAT + col] = (h != 0.f) ? 0x38 : 0x00;
        t[ty + 8 * r][tx] = (_Float16)h;
    }
    __syncthreads();
    #pragma unroll
    for (int r = 0; r < 4; ++r) {
        const int row = by + tx;
        const int col = bx + ty + 8 * r;
        HfT[(size_t)col * NMAT + row] = t[tx][ty + 8 * r];
    }
}

__global__ void initV(const float* __restrict__ llr, float* __restrict__ V) {
    const int col = blockIdx.x * blockDim.x + threadIdx.x;
    V[(size_t)blockIdx.y * NMAT + col] = llr[col];
}

__global__ __launch_bounds__(256) void signRowMin(const float* __restrict__ V,
                                                  const unsigned char* __restrict__ Hf8,
                                                  unsigned char* __restrict__ S8,
                                                  float* __restrict__ mag) {
    __shared__ float red[256];
    const int row = blockIdx.x;
    const size_t base = (size_t)row * NMAT;
    float m = 1e9f;
    for (int j = threadIdx.x; j < NMAT; j += 256) {
        const float v = V[base + j];
        S8[base + j] = (v > 0.f) ? 0x38 : ((v < 0.f) ? 0xB8 : 0x00);
        if (Hf8[base + j]) m = fminf(m, fabsf(v));
    }
    red[threadIdx.x] = m;
    __syncthreads();
    for (int s = 128; s > 0; s >>= 1) {
        if (threadIdx.x < s)
            red[threadIdx.x] = fminf(red[threadIdx.x], red[threadIdx.x + s]);
        __syncthreads();
    }
    if (threadIdx.x == 0) mag[row] = red[0];
}

__device__ __forceinline__ v16i loadFragA8(const unsigned char* rowBase, int half) {
    union { v16i v; unsigned long long q[8]; } u;
    #pragma unroll
    for (int r = 0; r < 8; ++r) {
        const int k = ((r >> 2) << 6) + ((r & 3) << 4) + (half << 3);
        u.q[r] = *(const unsigned long long*)(rowBase + k);
    }
    return u.v;
}
__device__ __forceinline__ v16i loadFragB8(const unsigned char* rowBase, int half) {
    union { v16i v; int4 x[4]; } u;
    #pragma unroll
    for (int g = 0; g < 4; ++g)
        u.x[g] = *(const int4*)(rowBase + (g << 5) + (half << 4));
    return u.v;
}

__global__ __launch_bounds__(256) void gemmCheck(const unsigned char* __restrict__ Hf8,
                                                 const unsigned char* __restrict__ S8,
                                                 const float* __restrict__ mag,
                                                 const float* __restrict__ gamma,
                                                 _Float16* __restrict__ c2vT) {
    constexpr int BM = 128, BN = 128, BK = 128, LDB = BK + 16;
    __shared__ unsigned char As[BM * LDB];
    __shared__ unsigned char Bs[BN * LDB];

    const int tid = threadIdx.x, lane = tid & 31, wave = tid >> 5;
    const int wm = wave >> 2, wn = wave & 3;
    const int m0 = blockIdx.y * BM, n0 = blockIdx.x * BN;
    const int half = lane >> 4, l16 = lane & 15;

    v8f acc[4][2] = {};

    for (int k0 = 0; k0 < NMAT; k0 += BK) {
        for (int c = tid; c < (BM * BK) / 16; c += 256) {
            const int row = c >> 3, off = (c & 7) << 4;
            copy16(Hf8 + (size_t)(m0 + row) * NMAT + k0 + off, &As[row * LDB + off]);
            copy16(S8  + (size_t)(n0 + row) * NMAT + k0 + off, &Bs[row * LDB + off]);
        }
        waitCopies();
        __syncthreads();

        v16i bf[2];
        #pragma unroll
        for (int c = 0; c < 2; ++c)
            bf[c] = loadFragA8(&Bs[(wn * 32 + c * 16 + l16) * LDB], half);
        #pragma unroll
        for (int r = 0; r < 4; ++r) {
            const v16i a = loadFragA8(&As[(wm * 64 + r * 16 + l16) * LDB], half);
            #pragma unroll
            for (int c = 0; c < 2; ++c)
                acc[r][c] = __builtin_amdgcn_wmma_f32_16x16x128_fp8_fp8(
                    a, bf[c], (short)0, acc[r][c], false, false);
        }
        __syncthreads();
    }

    const float g = gamma[0];
    #pragma unroll
    for (int r = 0; r < 4; ++r) {
        #pragma unroll
        for (int c = 0; c < 2; ++c) {
            const int col = n0 + wn * 32 + c * 16 + l16;
            const int rowBase = m0 + wm * 64 + r * 16 + half * 8;
            v8h out;
            #pragma unroll
            for (int v = 0; v < 8; ++v)
                out[v] = (_Float16)(g * signf(acc[r][c][v]) * mag[rowBase + v]);
            *(v8h*)(c2vT + (size_t)col * NMAT + rowBase) = out;
        }
    }
}

__global__ __launch_bounds__(256) void gemmVar(const _Float16* __restrict__ HfT,
                                               const _Float16* __restrict__ c2vT,
                                               const float* __restrict__ llr,
                                               float* __restrict__ V) {
    constexpr int BM = 128, BN = 128, BK = 64, LDH = BK + 8;
    __shared__ _Float16 As[BM * LDH];
    __shared__ _Float16 Bs[BN * LDH];

    const int tid = threadIdx.x, lane = tid & 31, wave = tid >> 5;
    const int wm = wave >> 2, wn = wave & 3;
    const int m0 = blockIdx.y * BM, n0 = blockIdx.x * BN;
    const int half = lane >> 4, l16 = lane & 15;

    v8f acc[4][2] = {};

    for (int k0 = 0; k0 < NMAT; k0 += BK) {
        for (int c = tid; c < (BM * BK * 2) / 16; c += 256) {
            const int row = c >> 3, offh = (c & 7) << 3;
            copy16(HfT  + (size_t)(m0 + row) * NMAT + k0 + offh, &As[row * LDH + offh]);
            copy16(c2vT + (size_t)(n0 + row) * NMAT + k0 + offh, &Bs[row * LDH + offh]);
        }
        waitCopies();
        __syncthreads();

        #pragma unroll
        for (int kk = 0; kk < BK; kk += 32) {
            v16h bf[2];
            #pragma unroll
            for (int c = 0; c < 2; ++c) {
                const _Float16* p = &Bs[(wn * 32 + c * 16 + l16) * LDH] + kk + half * 16;
                bf[c] = mk16(*(const v8h*)p, *(const v8h*)(p + 8));
            }
            #pragma unroll
            for (int r = 0; r < 4; ++r) {
                const _Float16* p = &As[(wm * 64 + r * 16 + l16) * LDH] + kk;
                const v16h a = mk16(*(const v8h*)(p + half * 16),
                                    *(const v8h*)(p + half * 16 + 8));
                #pragma unroll
                for (int c = 0; c < 2; ++c)
                    acc[r][c] = __builtin_amdgcn_wmma_f32_16x16x32_f16(
                        false, a, false, bf[c], (short)0, acc[r][c], false, false);
            }
        }
        __syncthreads();
    }

    #pragma unroll
    for (int r = 0; r < 4; ++r) {
        #pragma unroll
        for (int c = 0; c < 2; ++c) {
            const int col = n0 + wn * 32 + c * 16 + l16;
            const int rowBase = m0 + wm * 64 + r * 16 + half * 8;
            const float lj = llr[col];
            #pragma unroll
            for (int v = 0; v < 8; ++v)
                V[(size_t)(rowBase + v) * NMAT + col] = lj + acc[r][c][v];
        }
    }
}

extern "C" void kernel_launch(void* const* d_in, const int* in_sizes, int n_in,
                              void* d_out, int out_size, void* d_ws, size_t ws_size,
                              hipStream_t stream) {
    (void)in_sizes; (void)n_in; (void)out_size; (void)ws_size;
    const float* llr   = (const float*)d_in[0];
    const float* H     = (const float*)d_in[1];
    const float* gamma = (const float*)d_in[2];
    const int n_iter = 5;

    float* V = (float*)d_out;

    char* w = (char*)d_ws;
    const size_t NN = (size_t)NMAT * NMAT;
    _Float16*      c2vT = (_Float16*)w;                 w += NN * sizeof(_Float16);
    unsigned char* Hf8  = (unsigned char*)w;            w += NN;
    _Float16*      HfT  = (_Float16*)w;                 w += NN * sizeof(_Float16);
    unsigned char* S8   = (unsigned char*)w;            w += NN;
    float*         mag  = (float*)w;

    prepH<<<dim3(NMAT / 32, NMAT / 32), 256, 0, stream>>>(H, Hf8, HfT);
    initV<<<dim3(NMAT / 256, NMAT), 256, 0, stream>>>(llr, V);

    const dim3 gemmGrid(NMAT / 128, NMAT / 128);
    for (int it = 0; it < n_iter; ++it) {
        signRowMin<<<NMAT, 256, 0, stream>>>(V, Hf8, S8, mag);
        gemmCheck<<<gemmGrid, 256, 0, stream>>>(Hf8, S8, mag, gamma, c2vT);
        gemmVar<<<gemmGrid, 256, 0, stream>>>(HfT, c2vT, llr, V);
    }
}
